// SelfAttention_86629490360481
// MI455X (gfx1250) — compile-verified, matrix operand order checked by reading, not run on hardware
//
#include <hip/hip_runtime.h>

// ---------------------------------------------------------------------------
// Causal single-head attention, B=16, T=2048, C=384, DK=64.
// f32 inputs -> f16 Q/K/V -> WMMA f32 accumulation -> f32 output.
// Scale (1/sqrt(dk)) and log2(e) are folded into WQ, so attention scores come
// out of the QK^T WMMA already in log2 domain. Softmax row-sums are computed
// with an extra WMMA against a ones matrix (l' = a*l + P*1 has the same
// recurrence shape as O' = a*O + P*V), eliminating shuffle-based sum
// reductions entirely.
// ---------------------------------------------------------------------------

typedef __attribute__((ext_vector_type(16))) _Float16 v16h;
typedef __attribute__((ext_vector_type(8)))  _Float16 v8h;
typedef __attribute__((ext_vector_type(2)))  _Float16 v2h;
typedef __attribute__((ext_vector_type(8)))  float    v8f;
typedef __attribute__((ext_vector_type(4)))  float    v4f;

union frag16 {
    v16h v;
    v8h  h[2];
};

#define NB   16
#define NT   2048
#define NC   384
#define NDK  64

// scale * log2(e) = (1/8) * 1.4426950408889634
#define QSCALE 0.18033688011112043f

// ---------------------------------------------------------------------------
// Kernel 0: transpose + convert weights: W[c][n] (f32) -> Wt[w][n][c] (f16).
// WQ additionally scaled by 1/sqrt(dk)*log2(e) so QK^T is in log2 domain.
// ---------------------------------------------------------------------------
__global__ void wprep_kernel(const float* __restrict__ WQ,
                             const float* __restrict__ WK,
                             const float* __restrict__ WV,
                             _Float16* __restrict__ Wt)
{
    int idx = blockIdx.x * blockDim.x + threadIdx.x;
    if (idx >= 3 * NC * NDK) return;
    int w   = idx / (NC * NDK);
    int rem = idx % (NC * NDK);
    int c   = rem / NDK;
    int n   = rem % NDK;
    const float* W = (w == 0) ? WQ : ((w == 1) ? WK : WV);
    const float s  = (w == 0) ? QSCALE : 1.0f;
    Wt[(size_t)w * NDK * NC + (size_t)n * NC + c] = (_Float16)(W[c * NDK + n] * s);
}

// ---------------------------------------------------------------------------
// Kernel 1: projections. One wave computes one 16x16 output tile.
// M = B*T = 32768 rows, K = 384, N = 64 per weight.
// Outputs: Qh[row][dk], Kh[row][dk] (row-major f16), Vt[b][dk][t] (transposed).
// ---------------------------------------------------------------------------
__global__ __launch_bounds__(128) void proj_kernel(
    const float* __restrict__ x, const _Float16* __restrict__ Wt,
    _Float16* __restrict__ Qh, _Float16* __restrict__ Kh,
    _Float16* __restrict__ Vt)
{
    const int lane  = threadIdx.x & 31;
    const int wid   = threadIdx.x >> 5;
    const int gw    = blockIdx.x * 4 + wid;     // 0 .. 24575
    const int rt    = gw / 12;                  // row tile 0..2047
    const int sub   = gw % 12;
    const int w     = sub >> 2;                 // weight 0..2
    const int nt    = sub & 3;                  // n-tile 0..3
    const int lhalf = lane >> 4;
    const int l16   = lane & 15;
    const int kb    = lhalf * 8;                // per-lane k-run base

    const float*    xrow = x  + (size_t)(rt * 16 + l16) * NC;
    const _Float16* wrow = Wt + (size_t)w * NDK * NC + (size_t)(nt * 16 + l16) * NC;

    v8f acc = {};
#pragma unroll
    for (int ks = 0; ks < NC / 32; ++ks) {
        const int c0 = ks * 32 + kb;
        frag16 a, b;
        v4f f0 = *(const v4f*)(xrow + c0);
        v4f f1 = *(const v4f*)(xrow + c0 + 4);
        v4f f2 = *(const v4f*)(xrow + c0 + 16);
        v4f f3 = *(const v4f*)(xrow + c0 + 20);
#pragma unroll
        for (int i = 0; i < 4; ++i) {
            a.v[i]      = (_Float16)f0[i];
            a.v[4 + i]  = (_Float16)f1[i];
            a.v[8 + i]  = (_Float16)f2[i];
            a.v[12 + i] = (_Float16)f3[i];
        }
        b.h[0] = *(const v8h*)(wrow + c0);
        b.h[1] = *(const v8h*)(wrow + c0 + 16);
        acc = __builtin_amdgcn_wmma_f32_16x16x32_f16(false, a.v, false, b.v,
                                                     (short)0, acc, false, false);
    }

#pragma unroll
    for (int r = 0; r < 8; ++r) {
        const int orow = rt * 16 + r + lhalf * 8;
        const int col  = nt * 16 + l16;
        const _Float16 v = (_Float16)acc[r];
        if (w == 0)      Qh[(size_t)orow * NDK + col] = v;
        else if (w == 1) Kh[(size_t)orow * NDK + col] = v;
        else {
            const int bb = orow >> 11;          // T = 2048
            const int t  = orow & 2047;
            Vt[((size_t)bb * NDK + col) * NT + t] = v;
        }
    }
}

// ---------------------------------------------------------------------------
// Kernel 2: flash attention. One wave per 16-row query tile, 32 keys / step.
// ---------------------------------------------------------------------------
__global__ __launch_bounds__(128) void attn_kernel(
    const _Float16* __restrict__ Qh, const _Float16* __restrict__ Kh,
    const _Float16* __restrict__ Vt, float* __restrict__ out)
{
    __shared__ _Float16 lds_p[4][16 * 32];      // per-wave P tile (16x32)

    const int lane  = threadIdx.x & 31;
    const int wid   = threadIdx.x >> 5;
    const int gw    = blockIdx.x * 4 + wid;     // 0 .. 2047
    const int b     = gw >> 7;                  // 128 query tiles per batch
    const int q0    = (gw & 127) * 16;
    const int lhalf = lane >> 4;
    const int l16   = lane & 15;
    const int kb    = lhalf * 8;

    const _Float16* Qb = Qh + (size_t)b * NT * NDK;
    const _Float16* Kb = Kh + (size_t)b * NT * NDK;
    const _Float16* Vb = Vt + (size_t)b * NDK * NT;

    // Q fragments: dk 0..31 and dk 32..63 (pre-scaled by 1/sqrt(dk)*log2(e))
    frag16 aq0, aq1;
    {
        const _Float16* qrow = Qb + (size_t)(q0 + l16) * NDK;
        aq0.h[0] = *(const v8h*)(qrow + kb);
        aq0.h[1] = *(const v8h*)(qrow + kb + 16);
        aq1.h[0] = *(const v8h*)(qrow + 32 + kb);
        aq1.h[1] = *(const v8h*)(qrow + 32 + kb + 16);
    }

    // ones fragment for WMMA-based row sums
    frag16 ones;
#pragma unroll
    for (int i = 0; i < 16; ++i) ones.v[i] = (_Float16)1.0f;

    float m_i[8];
    v8f acc[4];                                 // O accumulator (16 x 64)
    v8f lacc = {};                              // row-sum accumulator
#pragma unroll
    for (int r = 0; r < 8; ++r) m_i[r] = -3.0e38f;
#pragma unroll
    for (int t = 0; t < 4; ++t) { v8f z = {}; acc[t] = z; }

    for (int j0 = 0; j0 < q0 + 16; j0 += 32) {
        // ---- S = Q K^T (two 16x16 tiles covering 32 keys); log2 domain ----
        frag16 bk00, bk01, bk10, bk11;
        const _Float16* krow0 = Kb + (size_t)(j0 + l16) * NDK;
        const _Float16* krow1 = Kb + (size_t)(j0 + 16 + l16) * NDK;
        bk00.h[0] = *(const v8h*)(krow0 + kb);
        bk00.h[1] = *(const v8h*)(krow0 + kb + 16);
        bk01.h[0] = *(const v8h*)(krow0 + 32 + kb);
        bk01.h[1] = *(const v8h*)(krow0 + 32 + kb + 16);
        bk10.h[0] = *(const v8h*)(krow1 + kb);
        bk10.h[1] = *(const v8h*)(krow1 + kb + 16);
        bk11.h[0] = *(const v8h*)(krow1 + 32 + kb);
        bk11.h[1] = *(const v8h*)(krow1 + 32 + kb + 16);

        v8f s0 = {}, s1 = {};
        s0 = __builtin_amdgcn_wmma_f32_16x16x32_f16(false, aq0.v, false, bk00.v, (short)0, s0, false, false);
        s0 = __builtin_amdgcn_wmma_f32_16x16x32_f16(false, aq1.v, false, bk01.v, (short)0, s0, false, false);
        s1 = __builtin_amdgcn_wmma_f32_16x16x32_f16(false, aq0.v, false, bk10.v, (short)0, s1, false, false);
        s1 = __builtin_amdgcn_wmma_f32_16x16x32_f16(false, aq1.v, false, bk11.v, (short)0, s1, false, false);

        // ---- causal mask: only the diagonal block needs per-element work ----
        if (j0 + 31 > q0) {
#pragma unroll
            for (int r = 0; r < 8; ++r) {
                const int qrow = q0 + r + lhalf * 8;
                if (j0 + l16 > qrow)      s0[r] = -1.0e30f;
                if (j0 + 16 + l16 > qrow) s1[r] = -1.0e30f;
            }
        }

        // ---- row max across 16 column lanes; 2 rows packed per f16x2 ----
        // (max is only a stabilization offset: it must be lane-consistent per
        //  row, not exact, so f16 rounding is harmless and halves shuffles)
        float mrow[8];
#pragma unroll
        for (int rr = 0; rr < 4; ++rr) {
            v2h hp;
            hp[0] = (_Float16)fmaxf(s0[2 * rr],     s1[2 * rr]);
            hp[1] = (_Float16)fmaxf(s0[2 * rr + 1], s1[2 * rr + 1]);
            int hi = __builtin_bit_cast(int, hp);
#pragma unroll
            for (int mlg = 0; mlg < 4; ++mlg) {
                int ot = __shfl_xor(hi, 1 << mlg, 32);
                v2h ho = __builtin_bit_cast(v2h, ot);
                v2h hc = __builtin_bit_cast(v2h, hi);
                v2h hm = __builtin_elementwise_max(hc, ho);
                hi = __builtin_bit_cast(int, hm);
            }
            v2h hr = __builtin_bit_cast(v2h, hi);
            mrow[2 * rr]     = (float)hr[0];
            mrow[2 * rr + 1] = (float)hr[1];
        }

        // ---- online softmax: p = exp2(s - m), alpha = exp2(m_old - m_new) ----
        float p0[8], p1[8], alpha[8];
#pragma unroll
        for (int r = 0; r < 8; ++r) {
            const float mn = fmaxf(m_i[r], mrow[r]);
            alpha[r] = __builtin_amdgcn_exp2f(m_i[r] - mn);
            p0[r]    = __builtin_amdgcn_exp2f(s0[r] - mn);
            p1[r]    = __builtin_amdgcn_exp2f(s1[r] - mn);
            m_i[r]   = mn;
        }
        // rescale O and row-sum accumulators (same recurrence)
#pragma unroll
        for (int t = 0; t < 4; ++t)
#pragma unroll
            for (int r = 0; r < 8; ++r) acc[t][r] *= alpha[r];
#pragma unroll
        for (int r = 0; r < 8; ++r) lacc[r] *= alpha[r];

        // ---- P: C/D layout -> A layout via LDS (wave-private tile) ----
        _Float16* pl = lds_p[wid];
#pragma unroll
        for (int r = 0; r < 8; ++r) {
            const int row = r + lhalf * 8;
            pl[row * 32 + l16]      = (_Float16)p0[r];
            pl[row * 32 + 16 + l16] = (_Float16)p1[r];
        }
        asm volatile("s_wait_dscnt 0x0" ::: "memory");

        frag16 ap;
        {
            const _Float16* prow = pl + l16 * 32;
            ap.h[0] = *(const v8h*)(prow + kb);
            ap.h[1] = *(const v8h*)(prow + kb + 16);
        }

        // ---- O += P V ; l += P 1  (V transposed: contiguous along keys) ----
#pragma unroll
        for (int t = 0; t < 4; ++t) {
            frag16 bv;
            const _Float16* vrow = Vb + (size_t)(t * 16 + l16) * NT + j0;
            bv.h[0] = *(const v8h*)(vrow + kb);
            bv.h[1] = *(const v8h*)(vrow + kb + 16);
            acc[t] = __builtin_amdgcn_wmma_f32_16x16x32_f16(false, ap.v, false, bv.v,
                                                            (short)0, acc[t], false, false);
        }
        lacc = __builtin_amdgcn_wmma_f32_16x16x32_f16(false, ap.v, false, ones.v,
                                                      (short)0, lacc, false, false);
    }

    // ---- epilogue: divide by row sum, store f32 ----
    float* ob = out + (size_t)b * NT * NDK;
#pragma unroll
    for (int r = 0; r < 8; ++r) {
        const float inv = 1.0f / lacc[r];
        const int   row = q0 + r + lhalf * 8;
#pragma unroll
        for (int t = 0; t < 4; ++t)
            ob[(size_t)row * NDK + t * 16 + l16] = acc[t][r] * inv;
    }
}

// ---------------------------------------------------------------------------
extern "C" void kernel_launch(void* const* d_in, const int* in_sizes, int n_in,
                              void* d_out, int out_size, void* d_ws, size_t ws_size,
                              hipStream_t stream) {
    const float* x  = (const float*)d_in[0];
    const float* WQ = (const float*)d_in[1];
    const float* WK = (const float*)d_in[2];
    const float* WV = (const float*)d_in[3];
    float* out = (float*)d_out;

    char* ws = (char*)d_ws;
    // Qh / Kh / Vt: each B*T*64 f16 = 4 MiB exactly. Wt: 3*64*384 f16 = 144 KiB.
    _Float16* Qh = (_Float16*)(ws);
    _Float16* Kh = (_Float16*)(ws + (size_t)4 * 1024 * 1024);
    _Float16* Vt = (_Float16*)(ws + (size_t)8 * 1024 * 1024);
    _Float16* Wt = (_Float16*)(ws + (size_t)12 * 1024 * 1024);

    // 1) weight transpose/convert (+ scale folding for WQ)
    {
        const int n = 3 * NC * NDK;
        wprep_kernel<<<(n + 255) / 256, 256, 0, stream>>>(WQ, WK, WV, Wt);
    }
    // 2) projections: 2048 row-tiles * 3 weights * 4 n-tiles = 24576 waves
    proj_kernel<<<24576 / 4, 128, 0, stream>>>(x, Wt, Qh, Kh, Vt);
    // 3) flash attention: 16 * 128 = 2048 query-tile waves
    attn_kernel<<<2048 / 4, 128, 0, stream>>>(Qh, Kh, Vt, out);
}
